// MultiHeadAttention_87875030876832
// MI455X (gfx1250) — matrix operand order repaired and checked by reading, not compiled, not run on hardware
//
#include <hip/hip_runtime.h>
#include <hip/hip_bf16.h>

// Problem constants (from reference)
#define BB  2
#define SS  2048
#define DD  1024
#define HH  16
#define DKK 64

typedef __attribute__((ext_vector_type(16))) __bf16        v16bf;
typedef __attribute__((ext_vector_type(8)))  float         v8f;
typedef __attribute__((ext_vector_type(8)))  unsigned int  v8u;
typedef __attribute__((ext_vector_type(4)))  unsigned int  u32x4;

// fp32 -> bf16 round-to-nearest-even
static __device__ __forceinline__ unsigned short f2bf(float f) {
  unsigned u = __builtin_bit_cast(unsigned, f);
  u += 0x7FFFu + ((u >> 16) & 1u);
  return (unsigned short)(u >> 16);
}
static __device__ __forceinline__ unsigned packbf(float lo, float hi) {
  return (unsigned)f2bf(lo) | ((unsigned)f2bf(hi) << 16);
}
static __device__ __forceinline__ u32x4 ld4(const unsigned short* p) {
  return *(const u32x4*)p;
}

// D = A(16x32 bf16) * B(32x16 bf16) + C(16x16 f32); fragments as two u32x4 each
static __device__ __forceinline__ v8f wmma_bf16(u32x4 a0, u32x4 a1, u32x4 b0, u32x4 b1, v8f c) {
  v8u av, bv;
#pragma unroll
  for (int i = 0; i < 4; ++i) { av[i] = a0[i]; av[i + 4] = a1[i]; bv[i] = b0[i]; bv[i + 4] = b1[i]; }
  return __builtin_amdgcn_wmma_f32_16x16x32_bf16(
      false, __builtin_bit_cast(v16bf, av),
      false, __builtin_bit_cast(v16bf, bv),
      (short)0, c, false, false);
}

// max-reduce across the 16 lanes of each half-wave (XOR butterfly, ds_swizzle imm)
static __device__ __forceinline__ float halfwave_max(float x) {
  x = fmaxf(x, __builtin_bit_cast(float, __builtin_amdgcn_ds_swizzle(__builtin_bit_cast(int, x), 0x041F)));
  x = fmaxf(x, __builtin_bit_cast(float, __builtin_amdgcn_ds_swizzle(__builtin_bit_cast(int, x), 0x081F)));
  x = fmaxf(x, __builtin_bit_cast(float, __builtin_amdgcn_ds_swizzle(__builtin_bit_cast(int, x), 0x101F)));
  x = fmaxf(x, __builtin_bit_cast(float, __builtin_amdgcn_ds_swizzle(__builtin_bit_cast(int, x), 0x201F)));
  return x;
}

// ---------------------------------------------------------------------------
// Kernel 0: one-time fp32 -> bf16 conversion (memory-bound, runs once)
// ---------------------------------------------------------------------------
__global__ __launch_bounds__(256)
void cvt_bf16_kernel(const float* __restrict__ in, unsigned short* __restrict__ out) {
  const size_t i = ((size_t)blockIdx.x * blockDim.x + threadIdx.x) * 4;
  float4 v = *(const float4*)(in + i);
  unsigned long long p = (unsigned long long)packbf(v.x, v.y) |
                         ((unsigned long long)packbf(v.z, v.w) << 32);
  *(unsigned long long*)(out + i) = p;
}

// ---------------------------------------------------------------------------
// Kernel 1: projection  Out = scale * (X @ W^T), all bf16.
// One wave computes a 16(M) x 64(N) tile; all fragments are b128 loads.
// MODE 0: store [B,H,S,DK]    MODE 1: store V transposed [B,H,DK,S]
// ---------------------------------------------------------------------------
template <int MODE>
__global__ __launch_bounds__(32)
void proj_kernel(const unsigned short* __restrict__ X, const unsigned short* __restrict__ W,
                 unsigned short* __restrict__ Out, float scale) {
  const int lane  = threadIdx.x;
  const int mbase = blockIdx.x * 16;
  const int nbase = blockIdx.y * 64;
  const int l15   = lane & 15;
  const int akoff = (lane < 16) ? 0 : 8;   // A-fragment K offset per half-wave
  const int bkoff = (lane < 16) ? 0 : 8;   // B-fragment K offset per half-wave
  const int mrow  = mbase + l15;

  v8f acc[4] = {};

  for (int kk = 0; kk < DD; kk += 32) {
    const unsigned short* arow = X + (size_t)mrow * DD + kk + akoff;
    u32x4 a0 = ld4(arow);       // K = kk+akoff    .. +7
    u32x4 a1 = ld4(arow + 16);  // K = kk+16+akoff .. +7
#pragma unroll
    for (int t = 0; t < 4; ++t) {
      const unsigned short* brow = W + (size_t)(nbase + t * 16 + l15) * DD + kk + bkoff;
      acc[t] = wmma_bf16(a0, a1, ld4(brow), ld4(brow + 16), acc[t]);
    }
  }

#pragma unroll
  for (int t = 0; t < 4; ++t) {
    const int n = nbase + t * 16 + l15;
    const int h = n >> 6, d = n & 63;
#pragma unroll
    for (int r = 0; r < 8; ++r) {
      const int m = mbase + r + akoff;
      const int bi = m >> 11, s = m & 2047;
      if (MODE == 0)
        Out[((size_t)(bi * HH + h) * SS + s) * DKK + d] = f2bf(acc[t][r] * scale);
      else  // V transposed: [B,H,DK,S]
        Out[((size_t)(bi * HH + h) * DKK + d) * SS + s] = f2bf(acc[t][r] * scale);
    }
  }
}

// ---------------------------------------------------------------------------
// One 32-key step for a 32-row query tile (two 16-row groups sharing the
// same K and V B-fragments -> 2x arithmetic intensity).
// o[g][0..3]: 16x64 output accumulators; o[g][4]: softmax denominator via
// the ones-column trick (P @ ones follows the same alpha-rescale recurrence).
// ---------------------------------------------------------------------------
template <bool MASKED>
static __device__ __forceinline__ void attn_step(
    int j0, int qbase, int l15, int akoff, int bkoff,
    const unsigned short* __restrict__ Kh, const unsigned short* __restrict__ Vh,
    unsigned short* __restrict__ Pw,                    // 2 x (16x32) bf16 tiles
    const u32x4 (&aq)[2][4], u32x4 ones,
    v8f (&o)[2][5], float (&mrow)[2][8]) {
  // --- K B-fragments for both 16-key tiles (shared by both row groups) ---
  u32x4 kf[2][4];
#pragma unroll
  for (int tt = 0; tt < 2; ++tt) {
    const unsigned short* krow = Kh + (size_t)(j0 + tt * 16 + l15) * DKK + bkoff;
    kf[tt][0] = ld4(krow);      kf[tt][1] = ld4(krow + 16);
    kf[tt][2] = ld4(krow + 32); kf[tt][3] = ld4(krow + 48);
  }

  // --- scores + online softmax per row group; P staged bf16 into LDS ---
#pragma unroll
  for (int g = 0; g < 2; ++g) {
    v8f sc[2];
#pragma unroll
    for (int tt = 0; tt < 2; ++tt) {
      v8f s = {};
      s = wmma_bf16(aq[g][0], aq[g][1], kf[tt][0], kf[tt][1], s);
      s = wmma_bf16(aq[g][2], aq[g][3], kf[tt][2], kf[tt][3], s);
      sc[tt] = s;
    }
    unsigned short* Pg = Pw + g * (16 * 32);
#pragma unroll
    for (int r = 0; r < 8; ++r) {
      float s0 = sc[0][r];
      float s1 = sc[1][r];
      if (MASKED) {
        const int q = qbase + g * 16 + r + akoff;
        if (j0 + l15 > q)      s0 = -3.0e38f;
        if (j0 + 16 + l15 > q) s1 = -3.0e38f;
      }
      const float rm = halfwave_max(fmaxf(s0, s1));
      const float mi = fmaxf(mrow[g][r], rm);
      const float alpha = exp2f((mrow[g][r] - mi) * 1.44269504f);
      mrow[g][r] = mi;
#pragma unroll
      for (int t = 0; t < 5; ++t) o[g][t][r] *= alpha;
      Pg[(r + akoff) * 32 + l15]      = f2bf(exp2f((s0 - mi) * 1.44269504f));
      Pg[(r + akoff) * 32 + 16 + l15] = f2bf(exp2f((s1 - mi) * 1.44269504f));
    }
  }

  asm volatile("s_wait_dscnt 0" ::: "memory");

  // --- V B-fragments (shared by both row groups) ---
  u32x4 vf[4][2];
#pragma unroll
  for (int t = 0; t < 4; ++t) {
    const unsigned short* vrow = Vh + (size_t)(t * 16 + l15) * SS + j0 + bkoff;
    vf[t][0] = ld4(vrow);
    vf[t][1] = ld4(vrow + 16);
  }

  // --- O += P @ V, L += P @ 1 per row group ---
#pragma unroll
  for (int g = 0; g < 2; ++g) {
    const unsigned short* prow = Pw + g * (16 * 32) + l15 * 32 + akoff;
    u32x4 pa0 = ld4(prow);
    u32x4 pa1 = ld4(prow + 16);
#pragma unroll
    for (int t = 0; t < 4; ++t)
      o[g][t] = wmma_bf16(pa0, pa1, vf[t][0], vf[t][1], o[g][t]);
    o[g][4] = wmma_bf16(pa0, pa1, ones, ones, o[g][4]);
  }
}

// ---------------------------------------------------------------------------
// Kernel 2: causal flash attention. Q/K bf16 [B,H,S,DK] (Q pre-scaled by
// 1/sqrt(DK)), V bf16 [B,H,DK,S]. Block = 8 waves; each wave owns a 32-row
// query tile (wave-private LDS, per-wave causal loop bound, no barriers).
// ---------------------------------------------------------------------------
__global__ __launch_bounds__(256)
void attn_kernel(const unsigned short* __restrict__ Qb,
                 const unsigned short* __restrict__ Kb,
                 const unsigned short* __restrict__ Vt,
                 unsigned short* __restrict__ Ob) {
  __shared__ unsigned short Plds[8 * 2 * 16 * 32];  // per-wave two 16x32 bf16 P tiles

  const int lane  = threadIdx.x & 31;
  const int wave  = threadIdx.x >> 5;
  const int bh    = blockIdx.y;
  const int qbase = blockIdx.x * 256 + wave * 32;   // multiple of 32

  const unsigned short* Qh = Qb + (size_t)bh * SS * DKK;
  const unsigned short* Kh = Kb + (size_t)bh * SS * DKK;
  const unsigned short* Vh = Vt + (size_t)bh * SS * DKK;  // [DK][S]
  unsigned short* Pw = Plds + wave * (2 * 16 * 32);

  const int l15   = lane & 15;
  const int akoff = (lane < 16) ? 0 : 8;
  const int bkoff = (lane < 16) ? 0 : 8;

  // Q A-fragments for both 16-row groups (K = 0..31 and 32..63 each)
  u32x4 aq[2][4];
#pragma unroll
  for (int g = 0; g < 2; ++g) {
    const unsigned short* qrow = Qh + (size_t)(qbase + g * 16 + l15) * DKK + akoff;
    aq[g][0] = ld4(qrow);
    aq[g][1] = ld4(qrow + 16);
    aq[g][2] = ld4(qrow + 32);
    aq[g][3] = ld4(qrow + 48);
  }

  const u32x4 ones = {0x3F803F80u, 0x3F803F80u, 0x3F803F80u, 0x3F803F80u};  // bf16 1.0 x8

  v8f o[2][5] = {};
  float mrow[2][8];
#pragma unroll
  for (int g = 0; g < 2; ++g)
#pragma unroll
    for (int r = 0; r < 8; ++r) mrow[g][r] = -3.0e38f;

  // steady state: all 32 keys strictly below qbase -> never masked
  int j0 = 0;
  for (; j0 + 32 <= qbase; j0 += 32) {
    __builtin_prefetch(Kh + (size_t)(j0 + 32) * DKK, 0, 1);
    attn_step<false>(j0, qbase, l15, akoff, bkoff, Kh, Vh, Pw, aq, ones, o, mrow);
  }
  // diagonal tile (j0 == qbase): keys qbase..qbase+31 straddle the diagonal
  attn_step<true>(j0, qbase, l15, akoff, bkoff, Kh, Vh, Pw, aq, ones, o, mrow);

  // --- finalize: O /= L (L = o[g][4], replicated across columns), store bf16 [B,S,D] ---
  const int bi = bh >> 4, h = bh & 15;
#pragma unroll
  for (int g = 0; g < 2; ++g)
#pragma unroll
    for (int r = 0; r < 8; ++r) {
      const float inv = 1.0f / o[g][4][r];
      const int srow = qbase + g * 16 + r + akoff;
      const size_t rowoff = ((size_t)bi * SS + srow) * DD + h * DKK;
#pragma unroll
      for (int t = 0; t < 4; ++t)
        Ob[rowoff + t * 16 + l15] = f2bf(o[g][t][r] * inv);
    }
}

// ---------------------------------------------------------------------------
// Kernel 3: output projection  out = Ob(bf16) @ Wo^T(bf16) + b_o  (fp32 out)
// ---------------------------------------------------------------------------
__global__ __launch_bounds__(32)
void out_proj_kernel(const unsigned short* __restrict__ A,
                     const unsigned short* __restrict__ W,
                     const float* __restrict__ bias,
                     float* __restrict__ Out) {
  const int lane  = threadIdx.x;
  const int mbase = blockIdx.x * 16;
  const int nbase = blockIdx.y * 64;
  const int l15   = lane & 15;
  const int akoff = (lane < 16) ? 0 : 8;
  const int bkoff = (lane < 16) ? 0 : 8;
  const int mrow  = mbase + l15;

  v8f acc[4] = {};

  for (int kk = 0; kk < DD; kk += 32) {
    const unsigned short* arow = A + (size_t)mrow * DD + kk + akoff;
    u32x4 a0 = ld4(arow);
    u32x4 a1 = ld4(arow + 16);
#pragma unroll
    for (int t = 0; t < 4; ++t) {
      const unsigned short* brow = W + (size_t)(nbase + t * 16 + l15) * DD + kk + bkoff;
      acc[t] = wmma_bf16(a0, a1, ld4(brow), ld4(brow + 16), acc[t]);
    }
  }

#pragma unroll
  for (int t = 0; t < 4; ++t) {
    const int n = nbase + t * 16 + l15;
    const float bb = bias[n];
#pragma unroll
    for (int r = 0; r < 8; ++r) {
      const int m = mbase + r + akoff;
      Out[(size_t)m * DD + n] = acc[t][r] + bb;
    }
  }
}

// ---------------------------------------------------------------------------
extern "C" void kernel_launch(void* const* d_in, const int* in_sizes, int n_in,
                              void* d_out, int out_size, void* d_ws, size_t ws_size,
                              hipStream_t stream) {
  const float* q  = (const float*)d_in[0];
  const float* k  = (const float*)d_in[1];
  const float* v  = (const float*)d_in[2];
  // d_in[3] = causal mask (handled analytically in-kernel)
  const float* wq = (const float*)d_in[4];
  const float* wk = (const float*)d_in[5];
  const float* wv = (const float*)d_in[6];
  const float* wo = (const float*)d_in[7];
  const float* bo = (const float*)d_in[8];
  float* out = (float*)d_out;

  unsigned short* ws = (unsigned short*)d_ws;
  const size_t NX = (size_t)BB * SS * DD;  // 4M elems (activations)
  const size_t NW = (size_t)DD * DD;       // 1M elems (weights)

  unsigned short* Xq = ws;                   // bf16 query input (8 MB)
  unsigned short* Xk = Xq + NX;
  unsigned short* Xv = Xk + NX;
  unsigned short* Wq = Xv + NX;              // bf16 weights (4 x 2 MB)
  unsigned short* Wk = Wq + NW;
  unsigned short* Wv = Wk + NW;
  unsigned short* Wo = Wv + NW;
  unsigned short* Qb = Wo + NW;              // bf16 Q (pre-scaled) [B,H,S,DK]
  unsigned short* Kb = Qb + NX;              // bf16 K  [B,H,S,DK]
  unsigned short* Vt = Kb + NX;              // bf16 V  [B,H,DK,S]
  unsigned short* Ob = Xq;                   // reuse Xq (dead after Q proj)

  // one-time fp32 -> bf16 conversion (each thread converts one float4)
  cvt_bf16_kernel<<<NX / 1024, 256, 0, stream>>>(q,  Xq);
  cvt_bf16_kernel<<<NX / 1024, 256, 0, stream>>>(k,  Xk);
  cvt_bf16_kernel<<<NX / 1024, 256, 0, stream>>>(v,  Xv);
  cvt_bf16_kernel<<<NW / 1024, 256, 0, stream>>>(wq, Wq);
  cvt_bf16_kernel<<<NW / 1024, 256, 0, stream>>>(wk, Wk);
  cvt_bf16_kernel<<<NW / 1024, 256, 0, stream>>>(wv, Wv);
  cvt_bf16_kernel<<<NW / 1024, 256, 0, stream>>>(wo, Wo);

  dim3 gproj(BB * SS / 16, DD / 64);
  proj_kernel<0><<<gproj, 32, 0, stream>>>(Xq, Wq, Qb, 0.125f);  // 1/sqrt(DK) folded in
  proj_kernel<0><<<gproj, 32, 0, stream>>>(Xk, Wk, Kb, 1.0f);
  proj_kernel<1><<<gproj, 32, 0, stream>>>(Xv, Wv, Vt, 1.0f);

  dim3 gattn(SS / 256, BB * HH);
  attn_kernel<<<gattn, 256, 0, stream>>>(Qb, Kb, Vt, Ob);

  dim3 gout(BB * SS / 16, DD / 64);
  out_proj_kernel<<<gout, 32, 0, stream>>>(Ob, Wo, bo, out);
}
